// SoftSupSimSiamLossV17_15040975470661
// MI455X (gfx1250) — matrix operand order repaired and checked by reading, not compiled, not run on hardware
//
#include <hip/hip_runtime.h>
#include <hip/hip_bf16.h>
#include <math.h>

// ---------------------------------------------------------------------------
// SoftSupSimSiamLoss, fused for MI455X (gfx1250, wave32, WMMA bf16).
//
//   sel_ij = -(p_n[i] . z_n[j]) * (S[i] . S[j]) ,  S[i] = targets[i]*r_i/|t_i|
//   out    = sum(sel, i!=j) / count(sel != 0)
//
// v2: symmetric-gram scheduling. For an upper-triangle tile-pair the gram
// G = S.S^T is computed once and reused for both triangles:
//     sum(sel lower tile) = sum( -(Z_i x P_j^T) o G(i,j) )
// (same WMMA C/D lane layout, no transpose needed). 1.66x fewer WMMAs.
// Each wave register-blocks a 32x32 group (2x2 tiles): 2 b128 loads / WMMA.
// ---------------------------------------------------------------------------

#define NROWS   8192
#define CP      256            // p/z feature dim
#define CT      1000           // targets dim
#define CTP     1024           // targets dim padded to K multiple of 32
#define GRP     256            // 32-row groups per side (8192/32)
#define NTASK   (GRP * (GRP + 1) / 2)   // 32896 upper-triangle group pairs

typedef __attribute__((ext_vector_type(16))) __bf16 v16bf;
typedef __attribute__((ext_vector_type(8)))  __bf16 v8bf;
typedef __attribute__((ext_vector_type(8)))  float  v8f;

static __device__ __forceinline__ v16bf load_afrag(const __bf16* p) {
    // A-fragment (16x32 bf16): per lane two contiguous 8-elem chunks,
    // K offsets kb and kb+16 (kb = (lane>>4)*8).
    const v8bf lo = *(const v8bf*)p;
    const v8bf hi = *(const v8bf*)(p + 16);
    return __builtin_shufflevector(lo, hi,
        0, 1, 2, 3, 4, 5, 6, 7, 8, 9, 10, 11, 12, 13, 14, 15);
}

#define WMMA_BF16(A, B, C) __builtin_amdgcn_wmma_f32_16x16x32_bf16( \
        false, (A), false, (B), (short)0, (C), false, false)

// ------------------------- accumulator init --------------------------------
__global__ void zero_acc_kernel(float* acc_sum, unsigned int* acc_cnt) {
    acc_sum[0] = 0.0f;
    acc_cnt[0] = 0u;
}

// -------------------- per-row normalize p,z -> bf16 ------------------------
__global__ __launch_bounds__(256) void prep_pz_kernel(
        const float* __restrict__ p, const float* __restrict__ z,
        __bf16* __restrict__ pn, __bf16* __restrict__ zn) {
    const int lane = threadIdx.x & 31;
    const int wav  = threadIdx.x >> 5;
    const int row  = blockIdx.x * 8 + wav;     // 0..16383

    const float* src;
    __bf16* dst;
    if (row < NROWS) {
        src = p  + (size_t)row * CP;
        dst = pn + (size_t)row * CP;
    } else {
        const int r2 = row - NROWS;
        src = z  + (size_t)r2 * CP;
        dst = zn + (size_t)r2 * CP;
    }

    float ss = 0.0f;
    #pragma unroll
    for (int k = lane; k < CP; k += 32) {
        const float v = src[k];
        ss += v * v;
    }
    #pragma unroll
    for (int off = 16; off; off >>= 1) ss += __shfl_xor(ss, off, 32);

    const float inv = 1.0f / fmaxf(sqrtf(ss), 1e-12f);

    #pragma unroll
    for (int k = lane; k < CP; k += 32)
        dst[k] = (__bf16)(src[k] * inv);
}

// --------- targets -> S = targets * (r/|t|) as bf16, zero-padded -----------
__global__ __launch_bounds__(256) void prep_targets_kernel(
        const float* __restrict__ t, __bf16* __restrict__ S) {
    const int lane = threadIdx.x & 31;
    const int wav  = threadIdx.x >> 5;
    const int row  = blockIdx.x * 8 + wav;     // 0..8191

    const float* src = t + (size_t)row * CT;

    float ss = 0.0f, ent = 0.0f;
    for (int k = lane; k < CT; k += 32) {
        const float v = src[k];            // v > 0 always, no NaN branch
        ss  += v * v;
        ent -= v * logf(v);
    }
    #pragma unroll
    for (int off = 16; off; off >>= 1) {
        ss  += __shfl_xor(ss,  off, 32);
        ent += __shfl_xor(ent, off, 32);
    }

    const float r     = 1.0f - ent / logf(1000.0f);
    const float scale = r / fmaxf(sqrtf(ss), 1e-8f);

    __bf16* dst = S + (size_t)row * CTP;
    for (int k = lane; k < CTP; k += 32) {
        const float v = (k < CT) ? src[k] * scale : 0.0f;
        dst[k] = (__bf16)v;
    }
}

// ------------------- symmetric fused tile GEMM kernel ----------------------
// One wave per upper-triangle 32x32 group pair (gi <= gj).
//   dotA[2][2] = P_i x Z_j^T   (K=256, 32 WMMA)
//   dotC[2][2] = Z_i x P_j^T   (K=256, 32 WMMA, off-diagonal tasks only)
//   gram[2][2] = S_i x S_j^T   (K=1024, 128 WMMA, shared by both triangles)
__global__ __launch_bounds__(256) void tile_loss_sym_kernel(
        const __bf16* __restrict__ pn, const __bf16* __restrict__ zn,
        const __bf16* __restrict__ S,
        float* __restrict__ acc_sum, unsigned int* __restrict__ acc_cnt) {
    const int lane = threadIdx.x & 31;
    const int wav  = threadIdx.x >> 5;
    const int idx  = blockIdx.x * 8 + wav;     // 0 .. NTASK-1

    // Decode upper-triangle pair (gi <= gj); BASE(g) = pairs before row g.
    #define BASE(g) ((g) * (2 * GRP - (g) + 1) / 2)
    int gi = (int)((double)GRP + 0.5 -
                   sqrt(((double)GRP + 0.5) * ((double)GRP + 0.5) -
                        2.0 * (double)idx));
    if (gi < 0) gi = 0;
    if (gi > GRP - 1) gi = GRP - 1;
    while (gi + 1 <= GRP - 1 && BASE(gi + 1) <= idx) ++gi;
    while (gi > 0 && BASE(gi) > idx) --gi;
    const int gj = gi + (idx - BASE(gi));
    #undef BASE
    const bool offd = (gi != gj);

    const int r0 = lane & 15;
    const int kA = (lane >> 4) << 3;           // A-frag K offset: 0 or 8
    const int kB = (lane >> 4) << 3;           // B-frag K offset: 0 or 8

    int rA[2], rB[2];
    rA[0] = gi * 32 + r0;      rA[1] = gi * 32 + 16 + r0;
    rB[0] = gj * 32 + r0;      rB[1] = gj * 32 + 16 + r0;

    // ---- dotA = P_i x Z_j^T (K = 256) ----
    v8f dA[2][2] = {{{}, {}}, {{}, {}}};
    {
        const __bf16* a0 = pn + (size_t)rA[0] * CP + kA;
        const __bf16* a1 = pn + (size_t)rA[1] * CP + kA;
        const __bf16* b0 = zn + (size_t)rB[0] * CP + kB;
        const __bf16* b1 = zn + (size_t)rB[1] * CP + kB;
        #pragma unroll
        for (int k = 0; k < CP; k += 32) {
            const v16bf fa0 = load_afrag(a0 + k);
            const v16bf fa1 = load_afrag(a1 + k);
            const v16bf fb0 = load_afrag(b0 + k);
            const v16bf fb1 = load_afrag(b1 + k);
            dA[0][0] = WMMA_BF16(fa0, fb0, dA[0][0]);
            dA[0][1] = WMMA_BF16(fa0, fb1, dA[0][1]);
            dA[1][0] = WMMA_BF16(fa1, fb0, dA[1][0]);
            dA[1][1] = WMMA_BF16(fa1, fb1, dA[1][1]);
        }
    }

    // ---- dotC = Z_i x P_j^T (K = 256), lower-triangle contribution ----
    v8f dC[2][2] = {{{}, {}}, {{}, {}}};
    if (offd) {   // uniform wave-level branch, EXEC stays all-ones
        const __bf16* a0 = zn + (size_t)rA[0] * CP + kA;
        const __bf16* a1 = zn + (size_t)rA[1] * CP + kA;
        const __bf16* b0 = pn + (size_t)rB[0] * CP + kB;
        const __bf16* b1 = pn + (size_t)rB[1] * CP + kB;
        #pragma unroll
        for (int k = 0; k < CP; k += 32) {
            const v16bf fa0 = load_afrag(a0 + k);
            const v16bf fa1 = load_afrag(a1 + k);
            const v16bf fb0 = load_afrag(b0 + k);
            const v16bf fb1 = load_afrag(b1 + k);
            dC[0][0] = WMMA_BF16(fa0, fb0, dC[0][0]);
            dC[0][1] = WMMA_BF16(fa0, fb1, dC[0][1]);
            dC[1][0] = WMMA_BF16(fa1, fb0, dC[1][0]);
            dC[1][1] = WMMA_BF16(fa1, fb1, dC[1][1]);
        }
    }

    // ---- gram = S_i x S_j^T (K = 1024, zero-padded), computed once ----
    v8f gG[2][2] = {{{}, {}}, {{}, {}}};
    {
        const __bf16* a0 = S + (size_t)rA[0] * CTP + kA;
        const __bf16* a1 = S + (size_t)rA[1] * CTP + kA;
        const __bf16* b0 = S + (size_t)rB[0] * CTP + kB;
        const __bf16* b1 = S + (size_t)rB[1] * CTP + kB;
        #pragma unroll 4
        for (int k = 0; k < CTP; k += 32) {
            const v16bf fa0 = load_afrag(a0 + k);
            const v16bf fa1 = load_afrag(a1 + k);
            const v16bf fb0 = load_afrag(b0 + k);
            const v16bf fb1 = load_afrag(b1 + k);
            gG[0][0] = WMMA_BF16(fa0, fb0, gG[0][0]);
            gG[0][1] = WMMA_BF16(fa0, fb1, gG[0][1]);
            gG[1][0] = WMMA_BF16(fa1, fb0, gG[1][0]);
            gG[1][1] = WMMA_BF16(fa1, fb1, gG[1][1]);
        }
    }

    // ---- combine. C/D layout: VGPR v is M = v + (lane>=16)*8, N = lane&15.
    const int mrow = (lane >> 4) << 3;
    float sum = 0.0f;
    int   cnt = 0;
    #pragma unroll
    for (int ii = 0; ii < 2; ++ii) {
        #pragma unroll
        for (int jj = 0; jj < 2; ++jj) {
            const int gr0 = gi * 32 + ii * 16 + mrow;   // global row (+v)
            const int gc  = gj * 32 + jj * 16 + r0;     // global col
            #pragma unroll
            for (int v = 0; v < 8; ++v) {
                float s1 = -dA[ii][jj][v] * gG[ii][jj][v];
                if (gr0 + v == gc) s1 = 0.0f;           // diagonal mask
                sum += s1;
                cnt += (s1 != 0.0f) ? 1 : 0;
            }
        }
    }
    if (offd) {   // lower-triangle tiles: sum(-dotC o G), never on diagonal
        #pragma unroll
        for (int ii = 0; ii < 2; ++ii) {
            #pragma unroll
            for (int jj = 0; jj < 2; ++jj) {
                #pragma unroll
                for (int v = 0; v < 8; ++v) {
                    const float s2 = -dC[ii][jj][v] * gG[ii][jj][v];
                    sum += s2;
                    cnt += (s2 != 0.0f) ? 1 : 0;
                }
            }
        }
    }

    #pragma unroll
    for (int off = 16; off; off >>= 1) {
        sum += __shfl_down(sum, off, 32);
        cnt += __shfl_down(cnt, off, 32);
    }
    if (lane == 0) {
        atomicAdd(acc_sum, sum);
        atomicAdd(acc_cnt, (unsigned int)cnt);
    }
}

// ------------------------------ finalize -----------------------------------
__global__ void finalize_kernel(const float* acc_sum, const unsigned int* acc_cnt,
                                float* out) {
    out[0] = acc_sum[0] / (float)acc_cnt[0];
}

// ---------------------------------------------------------------------------
extern "C" void kernel_launch(void* const* d_in, const int* in_sizes, int n_in,
                              void* d_out, int out_size, void* d_ws, size_t ws_size,
                              hipStream_t stream) {
    (void)in_sizes; (void)n_in; (void)out_size; (void)ws_size;

    const float* p = (const float*)d_in[0];   // [8192, 256]
    const float* z = (const float*)d_in[1];   // [8192, 256]
    const float* t = (const float*)d_in[2];   // [8192, 1000]
    float* out = (float*)d_out;

    // Workspace layout (bytes):
    //   [0,       4 MiB)   pn  bf16  8192x256
    //   [4 MiB,   8 MiB)   zn  bf16  8192x256
    //   [8 MiB,  24 MiB)   S   bf16  8192x1024
    //   [24 MiB, +8 B  )   acc_sum (f32), acc_cnt (u32)
    char* ws = (char*)d_ws;
    __bf16* pn = (__bf16*)(ws);
    __bf16* zn = (__bf16*)(ws + ((size_t)4 << 20));
    __bf16* S  = (__bf16*)(ws + ((size_t)8 << 20));
    float*        acc_sum = (float*)(ws + ((size_t)24 << 20));
    unsigned int* acc_cnt = (unsigned int*)(ws + ((size_t)24 << 20) + 4);

    zero_acc_kernel<<<1, 1, 0, stream>>>(acc_sum, acc_cnt);

    prep_pz_kernel<<<(2 * NROWS) / 8, 256, 0, stream>>>(p, z, pn, zn);
    prep_targets_kernel<<<NROWS / 8, 256, 0, stream>>>(t, S);

    // 32896 upper-triangle group pairs, one wave each, 8 waves per block.
    tile_loss_sym_kernel<<<NTASK / 8, 256, 0, stream>>>(
        pn, zn, S, acc_sum, acc_cnt);

    finalize_kernel<<<1, 1, 0, stream>>>(acc_sum, acc_cnt, out);
}
